// LSTMClassifier_17841294147989
// MI455X (gfx1250) — matrix operand order repaired and checked by reading, not compiled, not run on hardware
//
#include <hip/hip_runtime.h>

// ---------------------------------------------------------------------------
// Types for CDNA5 WMMA (wave32)
// ---------------------------------------------------------------------------
typedef __bf16 bf16_t;
typedef __attribute__((ext_vector_type(8)))  __bf16 v8bf;
typedef __attribute__((ext_vector_type(16))) __bf16 v16bf;
typedef __attribute__((ext_vector_type(8)))  float  v8f;

#define VOCAB 32000
#define EMB   512
#define HID   1024
#define BATCH 16
#define SEQ   256
#define ROWS  (BATCH * SEQ)   // 4096
#define GATES (4 * HID)       // 4096

// GEMM block tile
#define BM 128
#define BN 64
#define BK 32
#define ABYTES (BM * BK * 2)          // 8 KiB per buffer
#define BBYTES (BN * BK * 2)          // 4 KiB per buffer
#define LDS_TOTAL (2 * (ABYTES + BBYTES))   // 24 KiB

// ---------------------------------------------------------------------------
// CDNA5 async global -> LDS copy (ASYNCcnt-tracked)
// ---------------------------------------------------------------------------
__device__ __forceinline__ void async_load_b128(unsigned lds_off, const void* gptr) {
    asm volatile("global_load_async_to_lds_b128 %0, %1, off"
                 :: "v"(lds_off), "v"(gptr) : "memory");
}
__device__ __forceinline__ void wait_async_le6() {
    asm volatile("s_wait_asynccnt 0x6" ::: "memory");
}
__device__ __forceinline__ void wait_async_le0() {
    asm volatile("s_wait_asynccnt 0x0" ::: "memory");
}

// Copy nrows x 32 bf16 (row-major, leading dim ld) to LDS at byte offset lds_base.
// 16-byte chunks, 128 threads. nrows*4 chunks total.
__device__ __forceinline__ void stage_async(const bf16_t* gbase, int ld,
                                            int nrows, unsigned lds_base) {
#pragma unroll
    for (int c = threadIdx.x; c < nrows * 4; c += 128) {
        const int r  = c >> 2;
        const int kc = (c & 3) << 3;
        async_load_b128(lds_base + (unsigned)c * 16u,
                        (const void*)(gbase + (size_t)r * ld + kc));
    }
}

// ---------------------------------------------------------------------------
// WMMA fragment loaders (CDNA5 ISA 7.12.2 layouts), generic-pointer based
// (work for both global and LDS sources).
// A frag: rows m0..m0+15, cols k0..k0+31 of row-major [*, lda].
// Lanes 0-15: row=lane, K={0..7,16..23}; lanes 16-31: row=lane-16, K={8..15,24..31}.
// ---------------------------------------------------------------------------
__device__ __forceinline__ v16bf load_a_frag(const bf16_t* A, int lda, int m0, int k0) {
    const int lane = threadIdx.x & 31;
    const int row  = lane & 15;
    const int koff = (lane & 16) ? 8 : 0;
    const bf16_t* p = A + (size_t)(m0 + row) * lda + (k0 + koff);
    v8bf lo = *(const v8bf*)(p);
    v8bf hi = *(const v8bf*)(p + 16);
    v16bf a;
#pragma unroll
    for (int i = 0; i < 8; ++i) { a[i] = lo[i]; a[i + 8] = hi[i]; }
    return a;
}

// B frag: B(k,n) = W[n*ldw + k] (W row-major [N,K]); lanes 0-15: col=lane, K={0..7,16..23};
// lanes 16-31: col=lane-16, K={8..15,24..31}.
__device__ __forceinline__ v16bf load_b_frag(const bf16_t* W, int ldw, int n0, int k0) {
    const int lane = threadIdx.x & 31;
    const int col  = lane & 15;
    const int koff = (lane & 16) ? 8 : 0;
    const bf16_t* p = W + (size_t)(n0 + col) * ldw + (k0 + koff);
    v8bf lo = *(const v8bf*)(p);
    v8bf hi = *(const v8bf*)(p + 16);
    v16bf b;
#pragma unroll
    for (int i = 0; i < 8; ++i) { b[i] = lo[i]; b[i + 8] = hi[i]; }
    return b;
}

// C/D: lane&15 = column; VGPR v = row v (lanes 0-15) or v+8 (lanes 16-31).
__device__ __forceinline__ void store_c_bias(float* C, int ldc, int m0, int n0,
                                             v8f acc, const float* bias) {
    const int lane  = threadIdx.x & 31;
    const int col   = n0 + (lane & 15);
    const int mbase = m0 + ((lane & 16) ? 8 : 0);
    const float bv  = bias[col];
#pragma unroll
    for (int v = 0; v < 8; ++v)
        C[(size_t)(mbase + v) * ldc + col] = acc[v] + bv;
}

// ---------------------------------------------------------------------------
// WMMA GEMM:  C[M,N] = A[M,K](bf16) * W[N,K]^T(bf16) + bias[N]  (f32 out)
// 128x64 block tile, async-LDS double-buffered K pipeline, 4 waves,
// each wave owns two 16-row strips x four 16-col tiles (8 WMMAs / K-slab).
// ---------------------------------------------------------------------------
__global__ __launch_bounds__(128) void wmma_gemm_bias(const bf16_t* __restrict__ A,
                                                      const bf16_t* __restrict__ W,
                                                      const float*  __restrict__ bias,
                                                      float* __restrict__ C,
                                                      int K, int N) {
    extern __shared__ char smem[];
    const unsigned dynbase = __builtin_amdgcn_groupstaticsize();  // dynamic-LDS offset

    const int wave = threadIdx.x >> 5;
    const int m0   = blockIdx.y * BM;
    const int n0   = blockIdx.x * BN;

    // LDS byte offsets of the double buffers
    const unsigned aOff[2] = { dynbase,              dynbase + ABYTES };
    const unsigned bOff[2] = { dynbase + 2 * ABYTES, dynbase + 2 * ABYTES + BBYTES };

    // prologue: stage K-slab 0 into buffer 0 (6 async ops per wave)
    stage_async(A + (size_t)m0 * K, K, BM, aOff[0]);
    stage_async(W + (size_t)n0 * K, K, BN, bOff[0]);

    v8f acc[2][4] = {};
    int buf = 0;
    for (int k0 = 0; k0 < K; k0 += BK) {
        const bool has_next = (k0 + BK) < K;
        if (has_next) {  // stream next slab into the other buffer
            stage_async(A + (size_t)m0 * K + (k0 + BK), K, BM, aOff[buf ^ 1]);
            stage_async(W + (size_t)n0 * K + (k0 + BK), K, BN, bOff[buf ^ 1]);
            wait_async_le6();   // oldest 6 (this buf's slab) complete
        } else {
            wait_async_le0();
        }
        __syncthreads();

        const bf16_t* As = (const bf16_t*)(smem + (aOff[buf] - dynbase));
        const bf16_t* Bs = (const bf16_t*)(smem + (bOff[buf] - dynbase));

        v16bf a0 = load_a_frag(As, BK, 0 * 64 + wave * 16, 0);
        v16bf a1 = load_a_frag(As, BK, 1 * 64 + wave * 16, 0);
#pragma unroll
        for (int j = 0; j < 4; ++j) {
            v16bf b = load_b_frag(Bs, BK, j * 16, 0);
            acc[0][j] = __builtin_amdgcn_wmma_f32_16x16x32_bf16(
                false, a0, false, b, (short)0, acc[0][j], false, false);
            acc[1][j] = __builtin_amdgcn_wmma_f32_16x16x32_bf16(
                false, a1, false, b, (short)0, acc[1][j], false, false);
        }
        __syncthreads();   // protect buf before it is re-staged next iteration
        buf ^= 1;
    }

#pragma unroll
    for (int mi = 0; mi < 2; ++mi)
#pragma unroll
        for (int j = 0; j < 4; ++j)
            store_c_bias(C, N, m0 + mi * 64 + wave * 16, n0 + j * 16,
                         acc[mi][j], bias);
}

// ---------------------------------------------------------------------------
// Fused LSTM step. 64 blocks x 128 threads: wave g computes gate g's 16x16
// tile (K=HID, register double-buffered), gates exchanged via LDS for the
// pointwise cell update.
// ---------------------------------------------------------------------------
__device__ __forceinline__ float sigmoid_f(float x) {
    return 1.0f / (1.0f + __expf(-x));
}

__global__ __launch_bounds__(128) void lstm_step(const bf16_t* __restrict__ h_prev_bf,
                                                 const bf16_t* __restrict__ Whh_bf,
                                                 const float*  __restrict__ xg,
                                                 const float*  __restrict__ b_hh,
                                                 float*  __restrict__ c_state,
                                                 bf16_t* __restrict__ h_next_bf,
                                                 bf16_t* __restrict__ h_all_bf,
                                                 int t) {
    __shared__ float gbuf[4][16][16];

    const int j0   = blockIdx.x * 16;
    const int gate = threadIdx.x >> 5;
    const int nb   = gate * HID + j0;   // this wave's gate column block in W_hh

    // register-double-buffered K pipeline: loads for k+32 fly during WMMA on k
    v16bf a_c = load_a_frag(h_prev_bf, HID, 0, 0);
    v16bf b_c = load_b_frag(Whh_bf, HID, nb, 0);
    v8f acc = {};
    for (int k0 = 0; k0 < HID; k0 += 32) {
        const int kn = (k0 + 32 < HID) ? (k0 + 32) : 0;
        v16bf a_n = load_a_frag(h_prev_bf, HID, 0, kn);
        v16bf b_n = load_b_frag(Whh_bf, HID, nb, kn);
        acc = __builtin_amdgcn_wmma_f32_16x16x32_bf16(
            false, a_c, false, b_c, (short)0, acc, false, false);
        a_c = a_n;
        b_c = b_n;
    }

    // publish this wave's gate tile
    {
        const int lane  = threadIdx.x & 31;
        const int col   = lane & 15;
        const int mbase = (lane & 16) ? 8 : 0;
#pragma unroll
        for (int v = 0; v < 8; ++v)
            gbuf[gate][mbase + v][col] = acc[v];
    }
    __syncthreads();

    // fused pointwise LSTM update: 256 cells over 128 threads
#pragma unroll
    for (int cell = threadIdx.x; cell < 256; cell += 128) {
        const int m   = cell >> 4;       // batch index
        const int col = cell & 15;
        const int j   = j0 + col;
        const size_t xrow = ((size_t)m * SEQ + t) * (size_t)GATES;

        float gi = gbuf[0][m][col] + xg[xrow + 0 * HID + j] + b_hh[0 * HID + j];
        float gf = gbuf[1][m][col] + xg[xrow + 1 * HID + j] + b_hh[1 * HID + j];
        float gg = gbuf[2][m][col] + xg[xrow + 2 * HID + j] + b_hh[2 * HID + j];
        float go = gbuf[3][m][col] + xg[xrow + 3 * HID + j] + b_hh[3 * HID + j];

        float i_g = sigmoid_f(gi);
        float f_g = sigmoid_f(gf);
        float g_g = tanhf(gg);
        float o_g = sigmoid_f(go);

        const size_t ci = (size_t)m * HID + j;
        float c_new = f_g * c_state[ci] + i_g * g_g;
        float h     = o_g * tanhf(c_new);
        c_state[ci] = c_new;

        bf16_t hb = (bf16_t)h;
        h_next_bf[ci] = hb;
        h_all_bf[((size_t)m * SEQ + t) * (size_t)HID + j] = hb;
    }
}

// ---------------------------------------------------------------------------
// Helpers: embedding gather -> bf16, f32 -> bf16 convert, state init
// ---------------------------------------------------------------------------
__global__ __launch_bounds__(128) void embed_gather_bf16(const int* __restrict__ x,
                                                         const float* __restrict__ table,
                                                         bf16_t* __restrict__ e) {
    const int row = blockIdx.x;            // b*T + t
    const int idx = x[row];
    const float* src = table + (size_t)idx * EMB;
    bf16_t* dst = e + (size_t)row * EMB;
    for (int c = threadIdx.x * 4; c < EMB; c += blockDim.x * 4) {
        float4 v = *(const float4*)(src + c);
        dst[c + 0] = (bf16_t)v.x;
        dst[c + 1] = (bf16_t)v.y;
        dst[c + 2] = (bf16_t)v.z;
        dst[c + 3] = (bf16_t)v.w;
    }
}

__global__ __launch_bounds__(256) void cvt_f32_bf16(const float* __restrict__ src,
                                                    bf16_t* __restrict__ dst, size_t n) {
    size_t i = ((size_t)blockIdx.x * blockDim.x + threadIdx.x) * 4;
    const size_t stride = (size_t)gridDim.x * blockDim.x * 4;
    for (; i + 3 < n; i += stride) {
        float4 v = *(const float4*)(src + i);
        dst[i + 0] = (bf16_t)v.x;
        dst[i + 1] = (bf16_t)v.y;
        dst[i + 2] = (bf16_t)v.z;
        dst[i + 3] = (bf16_t)v.w;
    }
}

__global__ __launch_bounds__(256) void lstm_state_init(bf16_t* __restrict__ h0,
                                                       float* __restrict__ c) {
    const int i = blockIdx.x * blockDim.x + threadIdx.x;
    if (i < BATCH * HID) {
        h0[i] = (bf16_t)0.0f;
        c[i]  = 0.0f;
    }
}

// ---------------------------------------------------------------------------
// Launcher
// ---------------------------------------------------------------------------
extern "C" void kernel_launch(void* const* d_in, const int* in_sizes, int n_in,
                              void* d_out, int out_size, void* d_ws, size_t ws_size,
                              hipStream_t stream) {
    (void)in_sizes; (void)n_in; (void)out_size; (void)ws_size;

    const int*   x         = (const int*)d_in[0];
    const float* emb_table = (const float*)d_in[1];
    const float* W_ih      = (const float*)d_in[2];
    const float* W_hh      = (const float*)d_in[3];
    const float* b_ih      = (const float*)d_in[4];
    const float* b_hh      = (const float*)d_in[5];
    const float* head_W    = (const float*)d_in[6];
    const float* head_b    = (const float*)d_in[7];
    float* logits          = (float*)d_out;

    // Workspace carve-out (256B aligned)
    size_t off = 0;
    auto carve = [&](size_t bytes) -> void* {
        void* p = (char*)d_ws + off;
        off += (bytes + 255) & ~(size_t)255;
        return p;
    };
    bf16_t* e_bf     = (bf16_t*)carve((size_t)ROWS * EMB * 2);     //  4 MiB
    bf16_t* Wih_bf   = (bf16_t*)carve((size_t)GATES * EMB * 2);    //  4 MiB
    bf16_t* Whh_bf   = (bf16_t*)carve((size_t)GATES * HID * 2);    //  8 MiB
    bf16_t* headW_bf = (bf16_t*)carve((size_t)VOCAB * HID * 2);    // 62.5 MiB
    float*  xg       = (float*)carve((size_t)ROWS * GATES * 4);    // 64 MiB
    bf16_t* h_all_bf = (bf16_t*)carve((size_t)ROWS * HID * 2);     //  8 MiB
    bf16_t* h_ping   = (bf16_t*)carve((size_t)BATCH * HID * 2);
    bf16_t* h_pong   = (bf16_t*)carve((size_t)BATCH * HID * 2);
    float*  c_state  = (float*)carve((size_t)BATCH * HID * 4);

    // 1) zero h0 / c0 (workspace is poisoned, not zeroed)
    lstm_state_init<<<(BATCH * HID + 255) / 256, 256, 0, stream>>>(h_ping, c_state);

    // 2) weight conversions to bf16
    cvt_f32_bf16<<<1024, 256, 0, stream>>>(W_ih,   Wih_bf,   (size_t)GATES * EMB);
    cvt_f32_bf16<<<1024, 256, 0, stream>>>(W_hh,   Whh_bf,   (size_t)GATES * HID);
    cvt_f32_bf16<<<4096, 256, 0, stream>>>(head_W, headW_bf, (size_t)VOCAB * HID);

    // 3) embedding gather -> bf16
    embed_gather_bf16<<<ROWS, 128, 0, stream>>>(x, emb_table, e_bf);

    // 4) xg = e * W_ih^T + b_ih   [4096 x 4096]
    {
        dim3 grid(GATES / BN, ROWS / BM);
        wmma_gemm_bias<<<grid, 128, LDS_TOTAL, stream>>>(e_bf, Wih_bf, b_ih, xg,
                                                         EMB, GATES);
    }

    // 5) sequential recurrence: 256 dependent launches (stream order = barrier)
    for (int t = 0; t < SEQ; ++t) {
        bf16_t* hp = (t & 1) ? h_pong : h_ping;
        bf16_t* hn = (t & 1) ? h_ping : h_pong;
        lstm_step<<<HID / 16, 128, 0, stream>>>(hp, Whh_bf, xg, b_hh,
                                                c_state, hn, h_all_bf, t);
    }

    // 6) logits = h_all * head_W^T + head_b   [4096 x 32000]
    {
        dim3 grid(VOCAB / BN, ROWS / BM);
        wmma_gemm_bias<<<grid, 128, LDS_TOTAL, stream>>>(h_all_bf, headW_bf, head_b,
                                                         logits, HID, VOCAB);
    }
}
